// LSGModel_25469156065917
// MI455X (gfx1250) — matrix operand order repaired and checked by reading, not compiled, not run on hardware
//
#include <hip/hip_runtime.h>

typedef __attribute__((ext_vector_type(16))) __bf16          v16bf;
typedef __attribute__((ext_vector_type(16))) unsigned short  v16u;
typedef __attribute__((ext_vector_type(8)))  unsigned short  u16x8;
typedef __attribute__((ext_vector_type(8)))  float           v8f;

#define HW   65536     // 256*256 pixels
#define NSP  256       // superpixels
#define CPAD 128       // pf channels padded: 15 feat + 103 x + 10 zero

__device__ __forceinline__ unsigned short f2bf(float f) {
    union { float f; unsigned u; } v; v.f = f;
    unsigned r = v.u + 0x7fffu + ((v.u >> 16) & 1u);
    return (unsigned short)(r >> 16);
}
__device__ __forceinline__ float bf2f(unsigned short h) {
    union { unsigned u; float f; } v; v.u = ((unsigned)h) << 16;
    return v.f;
}
__device__ __forceinline__ v8f wmma_bf16(v16u a, v16u b, v8f c) {
    return __builtin_amdgcn_wmma_f32_16x16x32_bf16(
        false, __builtin_bit_cast(v16bf, a),
        false, __builtin_bit_cast(v16bf, b),
        (short)0, c, false, false);
}
// CDNA5 async global->LDS DMA (ASYNCcnt-tracked), CDNA5 ISA.
__device__ __forceinline__ void async_g2l_b128(unsigned lds_off, const void* gaddr) {
    asm volatile("global_load_async_to_lds_b128 %0, %1, off"
                 :: "v"(lds_off), "v"(gaddr) : "memory");
}
__device__ __forceinline__ void wait_async0() {
    asm volatile("s_wait_asynccnt 0" ::: "memory");
}

// ---------------- x (f32) -> pf rows 15..127 (bf16, rows >=118 zeroed) ------
__global__ void k_cvt_x(const float* __restrict__ x, unsigned short* __restrict__ pfx) {
    int idx = blockIdx.x * 256 + threadIdx.x;          // (128-15)*65536 threads
    int c = idx >> 16;
    pfx[idx] = (c < 103) ? f2bf(x[idx]) : (unsigned short)0;
}

// ---- fold 7/5/3 kernels into one bf16 weight, K padded to ic*64 + dy*8 + dx
__global__ void k_prep_w7(const float* __restrict__ k7, const float* __restrict__ k5,
                          const float* __restrict__ k3, unsigned short* __restrict__ w,
                          int IC) {
    int idx = blockIdx.x * 256 + threadIdx.x;          // 32*IC*64 threads
    if (idx >= 32 * IC * 64) return;
    int oc = idx / (IC * 64), rem = idx - oc * (IC * 64);
    int ic = rem >> 6, q = rem & 63;
    int dy = q >> 3, dx = q & 7;                       // 0..7 (row/col 7 = pad)
    float v = 0.f;
    if (dy < 7 && dx < 7) {
        v = k7[(oc * IC + ic) * 49 + dy * 7 + dx];
        int ey = dy - 1, ex = dx - 1;                  // 5x5 centered
        if ((unsigned)ey < 5u && (unsigned)ex < 5u) v += k5[(oc * IC + ic) * 25 + ey * 5 + ex];
        int ty = dy - 2, tx = dx - 2;                  // 3x3 centered
        if ((unsigned)ty < 3u && (unsigned)tx < 3u) v += k3[(oc * IC + ic) * 9 + ty * 3 + tx];
    }
    w[idx] = f2bf(v);
}

// ---------------- implicit-GEMM 7x7 conv, WMMA bf16 -> f32 ------------------
// Block: 8 waves = 2 oc-tiles x 4 px-tiles = 32 oc x 64 px (a quarter of one image row).
// LDS tile[IC][8][72] bf16 with zero-padded halo; k -> (ic,dy,dx) is bit-ops.
// Ping-pong double-buffered K-loop (Kp is always a multiple of 64) with
// explicitly vectorized fragment loads (2x b128 global + 2x b128 LDS each).
__global__ __launch_bounds__(256) void k_conv7(
        const unsigned short* __restrict__ w, const float* __restrict__ bias,
        const unsigned short* __restrict__ xin, float* __restrict__ y,
        int IC, int Kp /* = IC*64 */) {
    extern __shared__ char smem[];
    unsigned short* tile = (unsigned short*)smem;      // IC*576 elems
    const int tid = threadIdx.x, lane = tid & 31, wid = tid >> 5;
    const int pxBlk = blockIdx.x * 64;
    const int y0 = pxBlk >> 8, x0 = pxBlk & 255;
    // stage halo tile: branchless clamp + cndmask zero-fill
    const int total = IC * 576;
    for (int idx = tid; idx < total; idx += 256) {
        int ic = idx / 576, rem = idx - ic * 576;
        int dy = rem / 72, cc = rem - dy * 72;
        int yy = y0 + dy - 3, xx = x0 + cc - 3;
        int yc = min(max(yy, 0), 255), xc = min(max(xx, 0), 255);
        unsigned short v = xin[ic * HW + (yc << 8) + xc];
        if (yy != yc || xx != xc || dy == 7 || cc == 71) v = 0;
        tile[idx] = v;
    }
    __syncthreads();
    const int ocBase = (wid & 1) * 16;
    const int half = lane >> 4, mr = lane & 15, nc = lane & 15;
    const int pl = (wid >> 1) * 16 + nc;               // local px 0..63
    const unsigned short* wrow = w + (ocBase + mr) * Kp;

    auto loadA = [&](v16u& au, int k0) {
        u16x8 c0 = *(const u16x8*)(wrow + k0 + half * 16);
        u16x8 c1 = *(const u16x8*)(wrow + k0 + half * 16 + 8);
        #pragma unroll
        for (int j = 0; j < 8; ++j) { au[j] = c0[j]; au[8 + j] = c1[j]; }
    };
    auto loadB = [&](v16u& bu, int k0) {
        int kb = k0 + half * 16;                       // multiple of 16 -> kb&7==0
        const unsigned short* p0 =
            tile + ((kb >> 6) * 8 + ((kb >> 3) & 7)) * 72 + pl;
        u16x8 b0 = *(const u16x8*)(p0);                // k = kb+0..7   (row dy)
        u16x8 b1 = *(const u16x8*)(p0 + 72);           // k = kb+8..15  (row dy+1)
        #pragma unroll
        for (int j = 0; j < 8; ++j) { bu[j] = b0[j]; bu[8 + j] = b1[j]; }
    };

    v8f acc = {};
    v16u a0, b0, a1, b1;
    loadA(a0, 0); loadB(b0, 0);
    int k0 = 0;
    for (; k0 < Kp - 64; k0 += 64) {
        __builtin_prefetch(wrow + k0 + 128, 0, 3);     // global_prefetch_b8
        loadA(a1, k0 + 32); loadB(b1, k0 + 32);
        acc = wmma_bf16(a0, b0, acc);
        loadA(a0, k0 + 64); loadB(b0, k0 + 64);
        acc = wmma_bf16(a1, b1, acc);
    }
    loadA(a1, k0 + 32); loadB(b1, k0 + 32);
    acc = wmma_bf16(a0, b0, acc);
    acc = wmma_bf16(a1, b1, acc);

    const int n = pxBlk + pl;
    #pragma unroll
    for (int r = 0; r < 8; ++r) {
        int oc = ocBase + half * 8 + r;
        y[oc * HW + n] = acc[r] + bias[oc];
    }
}

// ---------------- BN batch stats + apply (ReLU, write bf16) -----------------
__global__ __launch_bounds__(256) void k_bn_stats(const float* __restrict__ y,
                                                  float* __restrict__ stats) {
    int ch = blockIdx.x;
    const float* row = y + ch * HW;
    float s = 0.f, s2 = 0.f;
    for (int i = threadIdx.x; i < HW; i += 256) { float v = row[i]; s += v; s2 += v * v; }
    __shared__ float sh[256], sh2[256];
    sh[threadIdx.x] = s; sh2[threadIdx.x] = s2; __syncthreads();
    for (int o = 128; o > 0; o >>= 1) {
        if (threadIdx.x < o) { sh[threadIdx.x] += sh[threadIdx.x + o]; sh2[threadIdx.x] += sh2[threadIdx.x + o]; }
        __syncthreads();
    }
    if (threadIdx.x == 0) { stats[ch * 2] = sh[0]; stats[ch * 2 + 1] = sh2[0]; }
}

__global__ void k_bn_apply(const float* __restrict__ y, const float* __restrict__ stats,
                           const float* __restrict__ g, const float* __restrict__ beta,
                           unsigned short* __restrict__ out) {
    int idx = blockIdx.x * 256 + threadIdx.x;          // 32*HW
    int ch = idx >> 16;
    float mu  = stats[ch * 2] * (1.f / HW);
    float var = stats[ch * 2 + 1] * (1.f / HW) - mu * mu;
    float v = (y[idx] - mu) * rsqrtf(var + 1e-5f) * g[ch] + beta[ch];
    out[idx] = f2bf(fmaxf(v, 0.f));
}

// ---------------- 1x1 conv (96 -> 15) via WMMA, writes pf rows 0..14 --------
__global__ void k_prep_w1(const float* __restrict__ ow, unsigned short* __restrict__ wf) {
    int idx = blockIdx.x * 256 + threadIdx.x;          // 16*96
    if (idx >= 16 * 96) return;
    wf[idx] = (idx / 96 < 15) ? f2bf(ow[idx]) : (unsigned short)0;
}

__global__ __launch_bounds__(256) void k_feat(
        const unsigned short* __restrict__ wf, const float* __restrict__ ob,
        const unsigned short* __restrict__ sb, unsigned short* __restrict__ pf) {
    const int lane = threadIdx.x & 31, wid = threadIdx.x >> 5;
    const int pxBase = blockIdx.x * 128 + wid * 16;
    const int half = lane >> 4, mr = lane & 15, nc = lane & 15;
    const int n = pxBase + nc;
    const unsigned short* wrow = wf + mr * 96;
    v8f acc = {};
    #pragma unroll
    for (int k0 = 0; k0 < 96; k0 += 32) {
        v16u au, bu;
        u16x8 c0 = *(const u16x8*)(wrow + k0 + half * 16);
        u16x8 c1 = *(const u16x8*)(wrow + k0 + half * 16 + 8);
        #pragma unroll
        for (int j = 0; j < 8; ++j) { au[j] = c0[j]; au[8 + j] = c1[j]; }
        #pragma unroll
        for (int i = 0; i < 16; ++i) bu[i] = sb[(k0 + half * 16 + i) * HW + n];
        acc = wmma_bf16(au, bu, acc);
    }
    #pragma unroll
    for (int r = 0; r < 8; ++r) {
        int m = half * 8 + r;
        if (m < 15) pf[m * HW + n] = f2bf(fmaxf(acc[r] + ob[m], 0.f));
    }
}

// ---------------- clustering prep -------------------------------------------
__global__ void k_pn2(const unsigned short* __restrict__ pf, float* __restrict__ pn2) {
    int n = blockIdx.x * 256 + threadIdx.x;
    float s = 0.f;
    for (int c = 0; c < CPAD; ++c) { float v = bf2f(pf[c * HW + n]); s += v * v; }
    pn2[n] = s;
}

__global__ void k_init_c(const unsigned short* __restrict__ pf, float* __restrict__ c) {
    int idx = blockIdx.x * 256 + threadIdx.x;          // 128*256
    int cc = idx >> 8, s = idx & 255;
    int gy = s >> 4, gx = s & 15;
    const unsigned short* base = pf + cc * HW + (gy * 16) * 256 + gx * 16;
    float sum = 0.f;
    for (int iy = 0; iy < 16; ++iy)
        for (int ix = 0; ix < 16; ++ix) sum += bf2f(base[iy * 256 + ix]);
    c[idx] = sum * (1.f / 256.f);
}

// c -> cb(bf16), cn2 = ||c||^2, zero accumulators. 1 block x 256.
__global__ void k_cn2cb(const float* __restrict__ c, unsigned short* __restrict__ cb,
                        float* __restrict__ cn2, float* __restrict__ cn_num,
                        float* __restrict__ cn_den) {
    int s = threadIdx.x;
    float sum = 0.f;
    for (int cc = 0; cc < CPAD; ++cc) {
        float v = c[cc * NSP + s];
        cb[cc * NSP + s] = f2bf(v);
        cn_num[cc * NSP + s] = 0.f;
        sum += v * v;
    }
    cn2[s] = sum;
    cn_den[s] = 0.f;
}

// ---------------- fused assign (dist -> softmax -> centroid accum) ----------
// LDS: pfT [128c][128p] bf16 @0 (32KB) + cbT [128c][256s] bf16 @32768 (64KB)
//      + distT [256s][128p] f32 @98304 (128KB) + red @229376 (2KB) = 226KB
// pfT/cbT staged with CDNA5 async global->LDS DMA (ASYNCcnt).
__global__ __launch_bounds__(256) void k_assign(
        const unsigned short* __restrict__ pf, const unsigned short* __restrict__ cb,
        const float* __restrict__ pn2, const float* __restrict__ cn2,
        float* __restrict__ cn_num, float* __restrict__ cn_den,
        float* __restrict__ qout, int writeQ) {
    extern __shared__ char smem[];
    unsigned short* pfT  = (unsigned short*)smem;             // [128][128]
    unsigned short* cbT  = (unsigned short*)(smem + 32768);   // [128][256]
    float* distT = (float*)(smem + 98304);                    // [256][128]
    float* redA  = (float*)(smem + 229376);                   // [256]
    float* redB  = redA + 256;
    const int tid = threadIdx.x, lane = tid & 31, wid = tid >> 5;
    const int pxBase = blockIdx.x * 128;
    const int half = lane >> 4, mr = lane & 15, nc = lane & 15;

    // async-stage pixel tile + centroid matrix into LDS (16B per lane per op)
    #pragma unroll
    for (int it = 0; it < 8; ++it) {                   // pfT: 2048 x 16B
        int idx = it * 256 + tid;
        int c = idx >> 4, chunk = idx & 15;
        async_g2l_b128(idx * 16, pf + c * HW + pxBase + chunk * 8);
    }
    #pragma unroll
    for (int it = 0; it < 16; ++it)                    // cbT: 4096 x 16B
        async_g2l_b128(32768 + (it * 256 + tid) * 16, cb + (it * 256 + tid) * 8);
    wait_async0();
    __syncthreads();

    // phase 1: score[s][p] = 2*dot - pn2 - cn2 via WMMA (A = c^T, B = pfT)
    for (int t = wid; t < 128; t += 8) {
        int sBase = (t >> 3) * 16, pBase = (t & 7) * 16;
        v8f acc = {};
        #pragma unroll
        for (int kk = 0; kk < 4; ++kk) {
            int k0 = kk * 32;
            v16u au, bu;
            #pragma unroll
            for (int i = 0; i < 16; ++i) {
                int ka = k0 + half * 16 + i;
                au[i] = cbT[ka * NSP + sBase + mr];
                bu[i] = pfT[(k0 + half * 16 + i) * 128 + pBase + nc];
            }
            acc = wmma_bf16(au, bu, acc);
        }
        #pragma unroll
        for (int r = 0; r < 8; ++r) {
            int s = sBase + half * 8 + r, p = pBase + nc;
            distT[s * 128 + p] = 2.f * acc[r] - pn2[pxBase + p] - cn2[s];
        }
    }
    __syncthreads();

    // phase 2: softmax over s per pixel, 2 threads/pixel; Q overwrites distT
    {
        int p = tid & 127, hs = tid >> 7;              // hs selects s half
        int s0 = hs * 128;
        float m = -3.4e38f;
        for (int s = s0; s < s0 + 128; ++s) m = fmaxf(m, distT[s * 128 + p]);
        redA[tid] = m;
        __syncthreads();
        float mm = fmaxf(redA[p], redA[128 + p]);
        float sum = 0.f;
        for (int s = s0; s < s0 + 128; ++s) sum += __expf(distT[s * 128 + p] - mm);
        redB[tid] = sum;
        __syncthreads();
        float inv = 1.f / (redB[p] + redB[128 + p]);
        for (int s = s0; s < s0 + 128; ++s) {
            float q = __expf(distT[s * 128 + p] - mm) * inv;
            distT[s * 128 + p] = q;
            if (writeQ) qout[s * HW + pxBase + p] = q;
        }
    }
    __syncthreads();
    if (writeQ) return;

    // phase 3: cn_num[c][s] += pfT (c x p) @ Q^T (p x s) via WMMA, f32 atomics
    for (int t = wid; t < 128; t += 8) {
        int cBase = (t >> 4) * 16, sBase = (t & 15) * 16;
        v8f acc = {};
        #pragma unroll
        for (int kk = 0; kk < 4; ++kk) {
            int k0 = kk * 32;
            v16u au, bu;
            const u16x8* arow = (const u16x8*)(pfT + (cBase + mr) * 128);
            u16x8 c0 = arow[(k0 + half * 16) >> 3];
            u16x8 c1 = arow[(k0 + half * 16 + 8) >> 3];
            #pragma unroll
            for (int j = 0; j < 8; ++j) { au[j] = c0[j]; au[8 + j] = c1[j]; }
            #pragma unroll
            for (int i = 0; i < 16; ++i)
                bu[i] = f2bf(distT[(sBase + nc) * 128 + k0 + half * 16 + i]);
            acc = wmma_bf16(au, bu, acc);
        }
        #pragma unroll
        for (int r = 0; r < 8; ++r)
            atomicAdd(&cn_num[(cBase + half * 8 + r) * NSP + sBase + nc], acc[r]);
    }
    {
        float sum = 0.f;
        int s = tid;                                   // 256 threads = 256 s
        for (int p = 0; p < 128; ++p) sum += distT[s * 128 + p];
        atomicAdd(&cn_den[s], sum);
    }
}

__global__ void k_update_c(const float* __restrict__ cn_num, const float* __restrict__ cn_den,
                           float* __restrict__ c) {
    int idx = blockIdx.x * 256 + threadIdx.x;          // 128*256
    c[idx] = cn_num[idx] / (cn_den[idx & 255] + 1e-8f);
}

// ============================================================================
extern "C" void kernel_launch(void* const* d_in, const int* in_sizes, int n_in,
                              void* d_out, int out_size, void* d_ws, size_t ws_size,
                              hipStream_t stream) {
    (void)in_sizes; (void)n_in; (void)out_size; (void)ws_size;
    const float* x      = (const float*)d_in[0];
    const float* s1k7   = (const float*)d_in[1];
    const float* s1k5   = (const float*)d_in[2];
    const float* s1k3   = (const float*)d_in[3];
    const float* s1b    = (const float*)d_in[4];
    const float* s1g    = (const float*)d_in[5];
    const float* s1beta = (const float*)d_in[6];
    const float* s2k7   = (const float*)d_in[7];
    const float* s2k5   = (const float*)d_in[8];
    const float* s2k3   = (const float*)d_in[9];
    const float* s2b    = (const float*)d_in[10];
    const float* s2g    = (const float*)d_in[11];
    const float* s2beta = (const float*)d_in[12];
    const float* s3k7   = (const float*)d_in[13];
    const float* s3k5   = (const float*)d_in[14];
    const float* s3k3   = (const float*)d_in[15];
    const float* s3b    = (const float*)d_in[16];
    const float* s3g    = (const float*)d_in[17];
    const float* s3beta = (const float*)d_in[18];
    const float* out_w  = (const float*)d_in[19];
    const float* out_b  = (const float*)d_in[20];
    float* out = (float*)d_out;

    // bump-allocate scratch (~38 MB)
    char* wsp = (char*)d_ws;
    size_t off = 0;
    auto alloc = [&](size_t bytes) -> void* {
        void* p = wsp + off;
        off += (bytes + 255) & ~(size_t)255;
        return p;
    };
    unsigned short* pf   = (unsigned short*)alloc((size_t)CPAD * HW * 2); // bf16 [128][HW]
    float*          ybuf = (float*)alloc((size_t)32 * HW * 4);            // conv f32 scratch
    unsigned short* sb   = (unsigned short*)alloc((size_t)96 * HW * 2);   // s1|s2|s3 bf16
    unsigned short* w7   = (unsigned short*)alloc((size_t)32 * 103 * 64 * 2);
    unsigned short* wf   = (unsigned short*)alloc((size_t)16 * 96 * 2);
    float*          stats= (float*)alloc(64 * 4);
    float*          pn2  = (float*)alloc((size_t)HW * 4);
    float*          cvec = (float*)alloc((size_t)CPAD * NSP * 4);
    unsigned short* cb   = (unsigned short*)alloc((size_t)CPAD * NSP * 2);
    float*          cn2  = (float*)alloc(NSP * 4);
    float*          cnum = (float*)alloc((size_t)CPAD * NSP * 4);
    float*          cden = (float*)alloc(NSP * 4);

    unsigned short* pfx = pf + (size_t)15 * HW;   // x lives at pf rows 15..117

    // 0) x -> bf16 into pf (rows >=118 zeroed)
    k_cvt_x<<<(128 - 15) * 256, 256, 0, stream>>>(x, pfx);

    // 1) stage s1: fold weights (K padded to IC*64), conv (LDS halo), BN
    k_prep_w7<<<(32 * 103 * 64) / 256, 256, 0, stream>>>(s1k7, s1k5, s1k3, w7, 103);
    k_conv7<<<1024, 256, 103 * 576 * 2, stream>>>(w7, s1b, pfx, ybuf, 103, 103 * 64);
    k_bn_stats<<<32, 256, 0, stream>>>(ybuf, stats);
    k_bn_apply<<<(32 * HW) / 256, 256, 0, stream>>>(ybuf, stats, s1g, s1beta, sb);

    // 2) stage s2
    k_prep_w7<<<(32 * 32 * 64) / 256, 256, 0, stream>>>(s2k7, s2k5, s2k3, w7, 32);
    k_conv7<<<1024, 256, 32 * 576 * 2, stream>>>(w7, s2b, sb, ybuf, 32, 32 * 64);
    k_bn_stats<<<32, 256, 0, stream>>>(ybuf, stats);
    k_bn_apply<<<(32 * HW) / 256, 256, 0, stream>>>(ybuf, stats, s2g, s2beta, sb + (size_t)32 * HW);

    // 3) stage s3 (input = concat(s1,s2) = sb rows 0..63)
    k_prep_w7<<<(32 * 64 * 64) / 256, 256, 0, stream>>>(s3k7, s3k5, s3k3, w7, 64);
    k_conv7<<<1024, 256, 64 * 576 * 2, stream>>>(w7, s3b, sb, ybuf, 64, 64 * 64);
    k_bn_stats<<<32, 256, 0, stream>>>(ybuf, stats);
    k_bn_apply<<<(32 * HW) / 256, 256, 0, stream>>>(ybuf, stats, s3g, s3beta, sb + (size_t)64 * HW);

    // 4) 1x1 conv -> pf rows 0..14
    k_prep_w1<<<6, 256, 0, stream>>>(out_w, wf);
    k_feat<<<512, 256, 0, stream>>>(wf, out_b, sb, pf);

    // 5) clustering
    k_pn2<<<HW / 256, 256, 0, stream>>>(pf, pn2);
    k_init_c<<<(CPAD * NSP) / 256, 256, 0, stream>>>(pf, cvec);

    const size_t shmem = 32768 + 65536 + 131072 + 2048;   // 226 KB
    for (int it = 0; it < 5; ++it) {
        k_cn2cb<<<1, 256, 0, stream>>>(cvec, cb, cn2, cnum, cden);
        k_assign<<<512, 256, shmem, stream>>>(pf, cb, pn2, cn2, cnum, cden, out, 0);
        k_update_c<<<(CPAD * NSP) / 256, 256, 0, stream>>>(cnum, cden, cvec);
    }
    k_cn2cb<<<1, 256, 0, stream>>>(cvec, cb, cn2, cnum, cden);
    k_assign<<<512, 256, shmem, stream>>>(pf, cb, pn2, cn2, cnum, cden, out, 1);
}
